// VQVAE_33208687133536
// MI455X (gfx1250) — matrix operand order repaired and checked by reading, not compiled, not run on hardware
//
#include <hip/hip_runtime.h>
#include <math.h>

// ---------------------------------------------------------------------------
// VQ-VAE forward for MI455X (gfx1250, wave32).
// All GEMMs on V_WMMA_F32_16X16X32_BF16 with hi/lo bf16 split-product
// (x*w ~= xh*wh + xh*wl + xl*wh, f32 accumulate) for near-fp32 accuracy at
// ~2.7x the f32-WMMA (16x16x4) rate. Weights are pre-split to padded bf16
// hi/lo arrays so the hot loop does zero conversion / guarding work.
// ---------------------------------------------------------------------------

typedef __bf16 v16bf __attribute__((ext_vector_type(16)));
typedef __bf16 v8bf  __attribute__((ext_vector_type(8)));
typedef float  v8f   __attribute__((ext_vector_type(8)));

#define B_ROWS   262144
#define LATENT   47
#define NCODES   512

#define ACT_NONE 0
#define ACT_RELU 1
#define ACT_ELU  2

__device__ __forceinline__ void split2(float x, __bf16& hi, __bf16& lo) {
  hi = (__bf16)x;                 // RTNE to bf16
  lo = (__bf16)(x - (float)hi);   // residual bits
}

__device__ __forceinline__ float act_apply(float v, int act) {
  if (act == ACT_RELU) return v > 0.0f ? v : 0.0f;
  if (act == ACT_ELU)  return v > 0.0f ? v : expm1f(v);
  return v;
}

__device__ __forceinline__ v16bf cat8(v8bf a, v8bf b) {
  return __builtin_shufflevector(a, b, 0,1,2,3,4,5,6,7,8,9,10,11,12,13,14,15);
}

__device__ __forceinline__ v8f wmma_bf16(v16bf a, v16bf b, v8f c) {
  return __builtin_amdgcn_wmma_f32_16x16x32_bf16(false, a, false, b,
                                                 (short)0, c, false, false);
}

// ---------------------------------------------------------------------------
// Pre-split kernels: fp32 -> padded, zero-filled bf16 hi/lo.
// ---------------------------------------------------------------------------
__global__ void split_w(const float* __restrict__ W, __bf16* __restrict__ WH,
                        __bf16* __restrict__ WL, int FI, int FO, int FIpad) {
  const int n = blockIdx.x;                 // grid = rowsPad
  for (int k = threadIdx.x; k < FIpad; k += blockDim.x) {
    float v = (n < FO && k < FI) ? W[(size_t)n * FI + k] : 0.0f;
    __bf16 hi, lo; split2(v, hi, lo);
    WH[(size_t)n * FIpad + k] = hi;
    WL[(size_t)n * FIpad + k] = lo;
  }
}

__global__ void split_emb(const float* __restrict__ EMB, __bf16* __restrict__ EH,
                          __bf16* __restrict__ EL, float* __restrict__ EN) {
  const int c = blockIdx.x;                 // grid = 512, block = 64
  const int k = threadIdx.x;                // K padded 47 -> 64
  float v = (k < LATENT) ? EMB[(size_t)c * LATENT + k] : 0.0f;
  __bf16 hi, lo; split2(v, hi, lo);
  EH[(size_t)c * 64 + k] = hi;
  EL[(size_t)c * 64 + k] = lo;
  if (k == 0) {
    const float* e = EMB + (size_t)c * LATENT;
    float s = 0.0f;
    for (int i = 0; i < LATENT; ++i) s += e[i] * e[i];
    EN[c] = s;
  }
}

// ---------------------------------------------------------------------------
// Fused GEMM: OUT[r, n] = act( sum_k X[r, k] * W[n, k] + bias[n] )
// W pre-split to bf16 hi/lo, padded [rowsPad=ntiles*16][FIpad], zero-filled.
// Block = 256 threads (8 waves = 2 M-groups x 4 N-groups), 64 rows/block.
// X (full K) staged once per block in dynamic LDS as hi/lo bf16.
// Per wave: 2 M-tiles x 1 N-tile at a time -> only 16 acc VGPRs live.
// ---------------------------------------------------------------------------
#define MBLK 64

__global__ __launch_bounds__(256)
void gemm_bias_act(const float* __restrict__ X,
                   const __bf16* __restrict__ WH, const __bf16* __restrict__ WL,
                   const float* __restrict__ bias, float* __restrict__ OUT,
                   int FI, int FIpad, int FO, int ldx, int ldo, int act)
{
  extern __shared__ __attribute__((aligned(16))) char smem[];
  const int XLD = FIpad + 8;                 // halves; rows stay 16B-aligned,
  __bf16* xh = (__bf16*)smem;                // bank-conflict-free frag reads
  __bf16* xl = xh + (size_t)MBLK * XLD;

  const int tid  = threadIdx.x;
  const int wave = tid >> 5;
  const int lane = tid & 31;
  const int wm = wave >> 2;                  // M-group (0..1)
  const int wn = wave & 3;                   // N-group (0..3)
  const int row0 = blockIdx.x * MBLK;
  const int ntiles  = (FO + 15) >> 4;
  const int kchunks = FIpad >> 5;

  // ---- stage X [64 x FIpad] once, hi/lo bf16, coalesced ----
  for (int r = wave; r < MBLK; r += 8) {
    const float* xp = X + (size_t)(row0 + r) * ldx;
    for (int k = lane; k < FIpad; k += 32) {
      float v = (k < FI) ? xp[k] : 0.0f;
      __bf16 hi, lo; split2(v, hi, lo);
      xh[r * XLD + k] = hi;
      xl[r * XLD + k] = lo;
    }
  }
  __syncthreads();

  const int arow  = lane & 15;
  const int akoff = (lane < 16) ? 0 : 8;     // A: k runs {akoff..+7, akoff+16..+23}
  const int bcol  = lane & 15;
  const int bkoff = (lane < 16) ? 0 : 8;     // B: k runs {bkoff..+7, bkoff+16..+23}
  const int mt0   = wm * 2;

  for (int i = 0; i < 6; ++i) {
    const int nt = wn + (i << 2);
    if (nt >= ntiles) break;                 // wave-uniform

    v8f acc0, acc1;
#pragma unroll
    for (int j = 0; j < 8; ++j) { acc0[j] = 0.0f; acc1[j] = 0.0f; }

    const size_t wrow = (size_t)(nt * 16 + bcol) * FIpad + bkoff;
    const int    ab0  = (mt0 * 16 + arow) * XLD + akoff;
    const int    ab1  = ab0 + 16 * XLD;

    for (int kc = 0; kc < kchunks; ++kc) {
      const int kb = kc << 5;
      const __bf16* wp = WH + wrow + kb;
      const __bf16* wq = WL + wrow + kb;
      v16bf bh = cat8(*(const v8bf*)wp, *(const v8bf*)(wp + 16));
      v16bf bl = cat8(*(const v8bf*)wq, *(const v8bf*)(wq + 16));

      v16bf ah0 = cat8(*(const v8bf*)(xh + ab0 + kb), *(const v8bf*)(xh + ab0 + kb + 16));
      v16bf al0 = cat8(*(const v8bf*)(xl + ab0 + kb), *(const v8bf*)(xl + ab0 + kb + 16));
      v16bf ah1 = cat8(*(const v8bf*)(xh + ab1 + kb), *(const v8bf*)(xh + ab1 + kb + 16));
      v16bf al1 = cat8(*(const v8bf*)(xl + ab1 + kb), *(const v8bf*)(xl + ab1 + kb + 16));

      acc0 = wmma_bf16(ah0, bh, acc0);
      acc0 = wmma_bf16(ah0, bl, acc0);
      acc0 = wmma_bf16(al0, bh, acc0);
      acc1 = wmma_bf16(ah1, bh, acc1);
      acc1 = wmma_bf16(ah1, bl, acc1);
      acc1 = wmma_bf16(al1, bh, acc1);
    }

    // epilogue: D layout vgpr j -> row j (lanes 0-15) / j+8 (lanes 16-31)
    const int n = nt * 16 + (lane & 15);
    if (n < FO) {
      const float bv = bias[n];
      const int rowadd = (lane < 16) ? 0 : 8;
#pragma unroll
      for (int j = 0; j < 8; ++j) {
        const int r = row0 + mt0 * 16 + j + rowadd;
        OUT[(size_t)r * ldo + n]        = act_apply(acc0[j] + bv, act);
        OUT[(size_t)(r + 16) * ldo + n] = act_apply(acc1[j] + bv, act);
      }
    }
  }
}

// ---------------------------------------------------------------------------
// VQ: argmin_k |z - e_k|^2 via WMMA scores (|e_k|^2 - 2 z.e_k); z_q is an
// exact fp32 gather; loss = 1.25 * mean(|q - z|^2).
// Block = 256 threads (8 waves), 32 rows/block, 128-code chunks; code
// fragments read straight from the pre-split global arrays (L2-resident).
// ---------------------------------------------------------------------------
#define VQ_MBLK 32
#define ZLD     72       // halves; 144B rows: aligned + conflict-free
#define G_LD    133

__global__ __launch_bounds__(256)
void vq_kernel(const float* __restrict__ Z, int ldz,
               const __bf16* __restrict__ EH, const __bf16* __restrict__ EL,
               const float* __restrict__ ENORM,
               const float* __restrict__ EMB,
               float* __restrict__ ZQ,            // row stride 47
               float* __restrict__ loss_acc)
{
  __shared__ alignas(16) __bf16 zh[VQ_MBLK * ZLD];
  __shared__ alignas(16) __bf16 zl[VQ_MBLK * ZLD];
  __shared__ float G[VQ_MBLK * G_LD];

  const int tid  = threadIdx.x;
  const int wave = tid >> 5;
  const int lane = tid & 31;
  const int row0 = blockIdx.x * VQ_MBLK;

  // ---- stage z [32 x 64] hi/lo (K padded 47 -> 64) ----
  for (int r = wave; r < VQ_MBLK; r += 8) {
    const float* zp = Z + (size_t)(row0 + r) * ldz;
    for (int k = lane; k < 64; k += 32) {
      float v = (k < LATENT) ? zp[k] : 0.0f;
      __bf16 hi, lo; split2(v, hi, lo);
      zh[r * ZLD + k] = hi;
      zl[r * ZLD + k] = lo;
    }
  }
  __syncthreads();

  const int arow  = lane & 15;
  const int akoff = (lane < 16) ? 0 : 8;
  const int bcol  = lane & 15;
  const int bkoff = (lane < 16) ? 0 : 8;

  float best_val = 3.4e38f;    // live in threads 0..31
  int   best_idx = 0;

  for (int cc = 0; cc < 4; ++cc) {
    const int nloc = wave * 16 + bcol;       // column inside this 128-chunk
    const int code = cc * 128 + nloc;        // global code index
    const float en = ENORM[code];

#pragma unroll
    for (int mt = 0; mt < 2; ++mt) {
      v8f acc;
#pragma unroll
      for (int j = 0; j < 8; ++j) acc[j] = 0.0f;
#pragma unroll
      for (int kc = 0; kc < 2; ++kc) {
        const size_t eo = (size_t)code * 64 + kc * 32 + bkoff;
        v16bf bh = cat8(*(const v8bf*)(EH + eo), *(const v8bf*)(EH + eo + 16));
        v16bf bl = cat8(*(const v8bf*)(EL + eo), *(const v8bf*)(EL + eo + 16));
        const int ab = (mt * 16 + arow) * ZLD + kc * 32 + akoff;
        v16bf ah = cat8(*(const v8bf*)(zh + ab), *(const v8bf*)(zh + ab + 16));
        v16bf al = cat8(*(const v8bf*)(zl + ab), *(const v8bf*)(zl + ab + 16));
        acc = wmma_bf16(ah, bh, acc);
        acc = wmma_bf16(ah, bl, acc);
        acc = wmma_bf16(al, bh, acc);
      }
      const int rowadd = (lane < 16) ? 0 : 8;
#pragma unroll
      for (int j = 0; j < 8; ++j) {
        const int r = mt * 16 + j + rowadd;
        G[r * G_LD + nloc] = en - 2.0f * acc[j];  // |z|^2 omitted (row-const)
      }
    }
    __syncthreads();

    if (tid < VQ_MBLK) {                     // per-row running argmin
      const float* g = G + tid * G_LD;
      for (int c = 0; c < 128; ++c) {
        float v = g[c];
        if (v < best_val) { best_val = v; best_idx = cc * 128 + c; }
      }
    }
    __syncthreads();                         // G reused next chunk
  }

  // ---- gather q (exact fp32), write z_q, accumulate loss ----
  float lsum = 0.0f;
  if (tid < VQ_MBLK) {
    const int rg = row0 + tid;
    const float* q = EMB + (size_t)best_idx * LATENT;
    const float* z = Z + (size_t)rg * ldz;
    float* out = ZQ + (size_t)rg * LATENT;
    for (int k = 0; k < LATENT; ++k) {
      float qv = q[k];
      out[k] = qv;
      float d = qv - z[k];
      lsum += d * d;
    }
  }
  if (wave == 0) {
    for (int m = 16; m >= 1; m >>= 1) lsum += __shfl_xor(lsum, m, 32);
    if (lane == 0) atomicAdd(loss_acc, lsum);
  }
}

__global__ void finalize_loss(const float* __restrict__ loss_acc,
                              float* __restrict__ out)
{
  // forward: q_loss == e_loss numerically -> loss = (1 + 0.25) * mean
  out[0] = 1.25f * loss_acc[0] / (float)((size_t)B_ROWS * LATENT);
}

// ---------------------------------------------------------------------------
extern "C" void kernel_launch(void* const* d_in, const int* in_sizes, int n_in,
                              void* d_out, int out_size, void* d_ws, size_t ws_size,
                              hipStream_t stream) {
  (void)in_sizes; (void)n_in; (void)out_size; (void)ws_size;
  const float* x   = (const float*)d_in[0];
  const float* emb = (const float*)d_in[17];
  const float* W[8]; const float* Bv[8];
  for (int i = 0; i < 8; ++i) { W[i] = (const float*)d_in[1 + 2 * i];
                                Bv[i] = (const float*)d_in[2 + 2 * i]; }

  float* out     = (float*)d_out;
  float* recon   = out;                                  // B x 375
  float* zq      = out + (size_t)B_ROWS * 375;           // B x 47
  float* lossout = out + (size_t)B_ROWS * 375 + (size_t)B_ROWS * 47;

  // ---- workspace layout ----
  float* buf0 = (float*)d_ws;                            // B x 256 f32
  float* buf1 = buf0 + (size_t)B_ROWS * 256;             // B x 160 f32
  float* lacc = buf1 + (size_t)B_ROWS * 160;             // 1 f32
  float* enorm_g = lacc + 1;                             // 512 f32
  size_t ofs = ((size_t)B_ROWS * 416 + 513) * sizeof(float);
  ofs = (ofs + 63) & ~(size_t)63;
  __bf16* sp = (__bf16*)((char*)d_ws + ofs);

  // per-layer dims: FI, FO -> FIpad (mult 32), rowsPad (mult 16)
  const int FIs[8]  = {375, 250, 150, 100,  47, 100, 150, 250};
  const int FOs[8]  = {250, 150, 100,  47, 100, 150, 250, 375};
  int FIpad[8], rowsPad[8];
  __bf16 *WHp[8], *WLp[8];
  for (int i = 0; i < 8; ++i) {
    FIpad[i]   = (FIs[i] + 31) & ~31;
    rowsPad[i] = (FOs[i] + 15) & ~15;
    WHp[i] = sp; sp += (size_t)rowsPad[i] * FIpad[i];
    WLp[i] = sp; sp += (size_t)rowsPad[i] * FIpad[i];
  }
  __bf16* eh_g = sp; sp += (size_t)NCODES * 64;
  __bf16* el_g = sp; sp += (size_t)NCODES * 64;

  hipMemsetAsync(lacc, 0, sizeof(float), stream);

  // ---- pre-split weights + emb (tiny: ~1.5 MB total) ----
  for (int i = 0; i < 8; ++i)
    split_w<<<dim3(rowsPad[i]), dim3(128), 0, stream>>>(
        W[i], WHp[i], WLp[i], FIs[i], FOs[i], FIpad[i]);
  split_emb<<<dim3(NCODES), dim3(64), 0, stream>>>(emb, eh_g, el_g, enorm_g);

  // ---- pipeline ----
  dim3 blk(256);
  dim3 grd(B_ROWS / MBLK);
  const float* xin[8] = {x, buf0, buf1, buf0, zq, buf0, buf1, buf0};
  float* xout[8]      = {buf0, buf1, buf0, buf1, buf0, buf1, buf0, recon};
  const int ldxs[8]   = {375, 256, 160, 112, 47, 112, 160, 256};
  const int ldos[8]   = {256, 160, 112, 64, 112, 160, 256, 375};
  const int acts[8]   = {ACT_RELU, ACT_ELU, ACT_ELU, ACT_NONE,
                         ACT_ELU, ACT_ELU, ACT_RELU, ACT_NONE};

  for (int i = 0; i < 8; ++i) {
    size_t shmem = (size_t)2 * MBLK * (FIpad[i] + 8) * sizeof(__bf16);
    gemm_bias_act<<<grd, blk, shmem, stream>>>(
        xin[i], WHp[i], WLp[i], Bv[i], xout[i],
        FIs[i], FIpad[i], FOs[i], ldxs[i], ldos[i], acts[i]);
    if (i == 3) {  // z_e ready in buf1 (stride 64) -> VQ
      vq_kernel<<<dim3(B_ROWS / VQ_MBLK), blk, 0, stream>>>(
          buf1, 64, eh_g, el_g, enorm_g, emb, zq, lacc);
    }
  }
  finalize_loss<<<1, 1, 0, stream>>>(lacc, lossout);
}
